// FFSNeRF_52132313039341
// MI455X (gfx1250) — matrix operand order repaired and checked by reading, not compiled, not run on hardware
//
#include <hip/hip_runtime.h>
#include <hip/hip_bf16.h>
#include <math.h>
#include <stdint.h>

#define R_    4096
#define NS_   64
#define NI_   64
#define NT_   128
#define HID_  128
#define NEARP 0.1f
#define FARP  4.0f

typedef __bf16 bf16t;
typedef __bf16 v16bf __attribute__((ext_vector_type(16)));
typedef float  v8f   __attribute__((ext_vector_type(8)));

__device__ __forceinline__ float bf2f(bf16t x) { return (float)x; }
__device__ __forceinline__ bf16t f2bf(float x) { return (bf16t)x; }
__device__ __forceinline__ int   imin(int a, int b) { return a < b ? a : b; }
__device__ __forceinline__ int   imax(int a, int b) { return a > b ? a : b; }
__device__ __forceinline__ float sigm(float x) { return 1.0f / (1.0f + __expf(-x)); }

// ---------------------------------------------------------------------------
// Pre-swizzled weight image (bf16 fragment-major + f32 biases), built once by
// FFS_prep into the workspace, then async-copied global->LDS by every block.
// Fragment layout: idx = ((kc*tilesN + nt)*32 + lane)*16 + e  (32 contiguous B
// per lane -> ds_load_b128 pairs when building WMMA B fragments).
//   B[k][n]: n = nt*16 + (lane&15), k = kc*32 + (lane>>4)*16 + e
// ---------------------------------------------------------------------------
// element offsets (bf16) inside the image
#define OFF_WS1F 0          // 1 kc x 8 nt   -> 4096
#define OFF_WS2F 4096       // 4 kc x 8 nt   -> 16384
#define OFF_WS3F 20480      // 4 kc x 1 nt   -> 2048
#define OFF_WS2T 22528      // 4 kc x 8 nt   -> 16384 (W2^T)
#define OFF_WS1T 38912      // 4 kc x 1 nt   -> 2048  (W1^T)
#define OFF_RS1F 40960      // 1 kc x 8 nt   -> 4096
#define OFF_RS2F 45056      // 4 kc x 8 nt   -> 16384
#define OFF_RS3F 61440      // 4 kc x 1 nt   -> 2048
#define IMG_BF16 63488      // total bf16 elements (126976 bytes)
#define BIAS_BYTE 126976    // f32 region: sb1(128) sb2(128) sb3p(16) rb1(128) rb2(128) rb3p(16)
#define BIAS_F32 544
#define IMG_BYTES 129152    // 126976 + 2176
#define IMG_PAD_BYTES 129536 // padded to 512B chunks
#define IMG_NCHUNK 253      // IMG_PAD_BYTES / 512

// bias f32 offsets within bias region
#define BOF_SB1 0
#define BOF_SB2 128
#define BOF_SB3 256
#define BOF_RB1 272
#define BOF_RB2 400
#define BOF_RB3 528

// ---------------------------------------------------------------------------
// Async global->LDS staging of the weight image (CDNA5 async-copy path).
// Each wave-instruction moves 32 lanes x 16B = 512B; tracked by ASYNCcnt.
// ---------------------------------------------------------------------------
__device__ __forceinline__ void stage_image(const bf16t* __restrict__ src,
                                            char* lds_base,
                                            int wave, int lane, int nwaves) {
  unsigned lds0 = (unsigned)(uintptr_t)lds_base + (unsigned)(lane * 16);
  for (int c = wave; c < IMG_NCHUNK; c += nwaves) {
    unsigned lds  = lds0 + (unsigned)(c * 512);
    unsigned voff = (unsigned)(c * 512 + lane * 16);
    asm volatile("global_load_async_to_lds_b128 %0, %1, %2"
                 :: "v"(lds), "v"(voff), "s"(src)
                 : "memory");
  }
  asm volatile("s_wait_asynccnt 0" ::: "memory");
}

// ---------------------------------------------------------------------------
// Wave-level bf16 WMMA GEMM: Y[M][N] = op(X[M][K] @ B + bias)
// X row-major in LDS; B pre-swizzled fragment-major in LDS.
// ---------------------------------------------------------------------------
__device__ __forceinline__ void mlp_layer(const bf16t* __restrict__ X, int K,
                                          const bf16t* __restrict__ Wf,
                                          const float* __restrict__ bias,
                                          bf16t* __restrict__ Y, int N,
                                          int tilesM, int tilesN,
                                          bool relu,
                                          int wave, int lane, int nwaves) {
  const int row = lane & 15;
  const int kh  = (lane >> 4) & 1;
  const int total = tilesM * tilesN;
  const int kchunks = K >> 5;
  for (int t = wave; t < total; t += nwaves) {
    const int m0 = (t / tilesN) * 16;
    const int nt = t % tilesN;
    v8f acc = {0.f, 0.f, 0.f, 0.f, 0.f, 0.f, 0.f, 0.f};
    for (int kc = 0; kc < kchunks; ++kc) {
      v16bf a, b;
      // A 16x32: lanes 0-15 -> K {0..15}; lanes 16-31 -> K {16..31}
      #pragma unroll
      for (int e = 0; e < 16; ++e) {
        const int ka = (kc << 5) + (kh << 4) + e;
        a[e] = X[(m0 + row) * K + ka];
      }
      // B: 32 contiguous bytes per lane from the pre-swizzled image
      const bf16t* bf = Wf + (((size_t)kc * tilesN + nt) * 32 + lane) * 16;
      #pragma unroll
      for (int e = 0; e < 16; ++e) b[e] = bf[e];
      acc = __builtin_amdgcn_wmma_f32_16x16x32_bf16(false, a, false, b,
                                                    (short)0, acc, false, false);
    }
    const int n  = nt * 16 + row;
    const float bv = bias ? bias[n] : 0.f;
    #pragma unroll
    for (int rr = 0; rr < 8; ++rr) {  // D: VGPR rr -> M = m0 + kh*8 + rr
      float v = acc[rr] + bv;
      if (relu) v = fmaxf(v, 0.f);
      Y[(m0 + (kh << 3) + rr) * N + n] = f2bf(v);
    }
  }
}

// ---------------------------------------------------------------------------
// Trilinear setup / eval (+ analytic derivative wrt fractional coords)
// ---------------------------------------------------------------------------
struct Tri {
  int   x0, y0, z0;
  float fx, fy, fz;
  float dcx, dcy, dcz;
};

__device__ __forceinline__ Tri tri_setup(float px, float py, float pz,
                                         int D, int H, int W) {
  Tri t;
  float qx = fminf(fmaxf(px, -1.f), 1.f);
  float qy = fminf(fmaxf(py, -1.f), 1.f);
  float qz = fminf(fmaxf(pz, -1.f), 1.f);
  float cx = (qx + 1.f) * 0.5f * (float)(D - 1);
  float cy = (qy + 1.f) * 0.5f * (float)(H - 1);
  float cz = (qz + 1.f) * 0.5f * (float)(W - 1);
  t.x0 = imin(imax((int)floorf(cx), 0), D - 2);
  t.y0 = imin(imax((int)floorf(cy), 0), H - 2);
  t.z0 = imin(imax((int)floorf(cz), 0), W - 2);
  t.fx = cx - (float)t.x0;
  t.fy = cy - (float)t.y0;
  t.fz = cz - (float)t.z0;
  t.dcx = (px > -1.f && px < 1.f) ? 0.5f * (float)(D - 1) : 0.f;
  t.dcy = (py > -1.f && py < 1.f) ? 0.5f * (float)(H - 1) : 0.f;
  t.dcz = (pz > -1.f && pz < 1.f) ? 0.5f * (float)(W - 1) : 0.f;
  return t;
}

template <bool GRAD, int NCH>
__device__ __forceinline__ void tri_eval(const float* __restrict__ g,
                                         int D, int H, int W, const Tri& t,
                                         float* feat, float* gx, float* gy, float* gz) {
  const long cs   = (long)D * (long)H * (long)W;
  const long base = ((long)t.x0 * (long)H + (long)t.y0) * (long)W + (long)t.z0;
  const int  sW   = W;
  const long sHW  = (long)H * (long)W;
  const float wz1 = t.fz, wz0 = 1.f - wz1;
  const float wy1 = t.fy, wy0 = 1.f - wy1;
  const float wx1 = t.fx, wx0 = 1.f - wx1;
  #pragma unroll
  for (int c = 0; c < NCH; ++c) {
    const float* p = g + (long)c * cs + base;
    float v000 = p[0],  v001 = p[1];
    float v010 = p[sW], v011 = p[sW + 1];
    const float* q = p + sHW;
    float v100 = q[0],  v101 = q[1];
    float v110 = q[sW], v111 = q[sW + 1];
    float c00 = v000 * wz0 + v001 * wz1;
    float c01 = v010 * wz0 + v011 * wz1;
    float c10 = v100 * wz0 + v101 * wz1;
    float c11 = v110 * wz0 + v111 * wz1;
    float c0 = c00 * wy0 + c01 * wy1;
    float c1 = c10 * wy0 + c11 * wy1;
    feat[c] = c0 * wx0 + c1 * wx1;
    if constexpr (GRAD) {
      gx[c] = c1 - c0;
      gy[c] = (c01 - c00) * wx0 + (c11 - c10) * wx1;
      gz[c] = ((v001 - v000) * wy0 + (v011 - v010) * wy1) * wx0 +
              ((v101 - v100) * wy0 + (v111 - v110) * wy1) * wx1;
    }
  }
}

// ===========================================================================
// Kernel 0: build the pre-swizzled bf16 weight image (+ biases) in workspace
// ===========================================================================
__device__ __forceinline__ void fill_fwd(bf16t* dst, const float* W,
                                         int realK, int realN, int K, int tilesN,
                                         int tid, int stride) {
  const int total = (K >> 5) * tilesN * 512;
  for (int i = tid; i < total; i += stride) {
    int e = i & 15, lane = (i >> 4) & 31, ti = i >> 9;
    int kc = ti / tilesN, nt = ti % tilesN;
    int n = nt * 16 + (lane & 15);
    int k = kc * 32 + ((lane >> 4) << 4) + e;
    float v = (k < realK && n < realN) ? W[k * realN + n] : 0.f;
    dst[i] = f2bf(v);
  }
}

__device__ __forceinline__ void fill_bT(bf16t* dst, const float* W,
                                        int realN, int ldw, int K, int tilesN,
                                        int tid, int stride) {
  const int total = (K >> 5) * tilesN * 512;
  for (int i = tid; i < total; i += stride) {
    int e = i & 15, lane = (i >> 4) & 31, ti = i >> 9;
    int kc = ti / tilesN, nt = ti % tilesN;
    int n = nt * 16 + (lane & 15);
    int k = kc * 32 + ((lane >> 4) << 4) + e;
    float v = (n < realN) ? W[n * ldw + k] : 0.f;   // B[k][n] = W^T
    dst[i] = f2bf(v);
  }
}

__global__ void __launch_bounds__(256)
FFS_prep(const float* __restrict__ sW1, const float* __restrict__ sb1,
         const float* __restrict__ sW2, const float* __restrict__ sb2,
         const float* __restrict__ sW3, const float* __restrict__ sb3,
         const float* __restrict__ rW1, const float* __restrict__ rb1,
         const float* __restrict__ rW2, const float* __restrict__ rb2,
         const float* __restrict__ rW3, const float* __restrict__ rb3,
         bf16t* __restrict__ wimg) {
  const int tid = blockIdx.x * blockDim.x + threadIdx.x;
  const int stride = gridDim.x * blockDim.x;
  fill_fwd(wimg + OFF_WS1F, sW1, 11, 128, 32, 8, tid, stride);
  fill_fwd(wimg + OFF_WS2F, sW2, 128, 128, 128, 8, tid, stride);
  fill_fwd(wimg + OFF_WS3F, sW3, 128, 1, 128, 1, tid, stride);
  fill_bT (wimg + OFF_WS2T, sW2, 128, 128, 128, 8, tid, stride);
  fill_bT (wimg + OFF_WS1T, sW1, 11, 128, 128, 1, tid, stride);
  fill_fwd(wimg + OFF_RS1F, rW1, 14, 128, 32, 8, tid, stride);
  fill_fwd(wimg + OFF_RS2F, rW2, 128, 128, 128, 8, tid, stride);
  fill_fwd(wimg + OFF_RS3F, rW3, 128, 3, 128, 1, tid, stride);
  float* bimg = (float*)((char*)wimg + BIAS_BYTE);
  for (int i = tid; i < 128; i += stride) {
    bimg[BOF_SB1 + i] = sb1[i];
    bimg[BOF_SB2 + i] = sb2[i];
    bimg[BOF_RB1 + i] = rb1[i];
    bimg[BOF_RB2 + i] = rb2[i];
  }
  for (int i = tid; i < 16; i += stride) {
    bimg[BOF_SB3 + i] = (i == 0) ? sb3[0] : 0.f;
    bimg[BOF_RB3 + i] = (i < 3) ? rb3[i] : 0.f;
  }
  // zero the 512B-alignment pad so the async copy is deterministic
  bf16t* pad = (bf16t*)((char*)wimg + IMG_BYTES);
  for (int i = tid; i < (IMG_PAD_BYTES - IMG_BYTES) / 2; i += stride) pad[i] = f2bf(0.f);
}

// ===========================================================================
// Kernel 1: coarse sdf (64 points / block, sdf MLP via WMMA)
// ===========================================================================
__global__ void __launch_bounds__(256)
FFS_coarse(const float* __restrict__ rays_o, const float* __restrict__ rays_d,
           const float* __restrict__ z_vals,
           const float* __restrict__ grid0, const float* __restrict__ grid1,
           const float* __restrict__ origin, const float* __restrict__ dims,
           const bf16t* __restrict__ wimg,
           float* __restrict__ sdf_c) {
  extern __shared__ __align__(16) char smem[];
  bf16t* img  = (bf16t*)smem;
  float* bimg = (float*)(smem + BIAS_BYTE);
  bf16t* ap   = (bf16t*)(smem + IMG_PAD_BYTES);
  bf16t* Xs = ap; ap += 64 * 32;
  bf16t* H1 = ap; ap += 64 * HID_;
  bf16t* H2 = ap; ap += 64 * HID_;
  bf16t* H3 = ap;

  const int tid = threadIdx.x;
  const int wave = tid >> 5, lane = tid & 31;
  stage_image(wimg, smem, wave, lane, 8);

  const int p = blockIdx.x * 64 + tid;
  int inMask = 0;
  if (tid < 64) {
    int ray = p / NS_;
    int s   = p - ray * NS_;
    float zc = z_vals[ray * NS_ + s];
    float px = rays_o[ray * 3 + 0] + rays_d[ray * 3 + 0] * zc;
    float py = rays_o[ray * 3 + 1] + rays_d[ray * 3 + 1] * zc;
    float pz = rays_o[ray * 3 + 2] + rays_d[ray * 3 + 2] * zc;
    float pnx = 2.f * (px - origin[0]) / dims[0] - 1.f;
    float pny = 2.f * (py - origin[1]) / dims[1] - 1.f;
    float pnz = 2.f * (pz - origin[2]) / dims[2] - 1.f;
    inMask = (fabsf(pnx) <= 1.f && fabsf(pny) <= 1.f && fabsf(pnz) <= 1.f) ? 1 : 0;
    Tri t0 = tri_setup(pnx, pny, pnz, 96, 96, 96);
    Tri t1 = tri_setup(pnx, pny, pnz, 192, 192, 192);
    __builtin_prefetch(grid0 + (((long)t0.x0 * 96 + t0.y0) * 96 + t0.z0), 0, 0);
    __builtin_prefetch(grid1 + (((long)t1.x0 * 192 + t1.y0) * 192 + t1.z0), 0, 0);
    float f0[8], f1[8];
    tri_eval<false, 8>(grid0, 96, 96, 96, t0, f0, nullptr, nullptr, nullptr);
    tri_eval<false, 8>(grid1, 192, 192, 192, t1, f1, nullptr, nullptr, nullptr);
    bf16t* xr = Xs + tid * 32;
    #pragma unroll
    for (int c = 0; c < 4; ++c) { xr[c] = f2bf(f0[c]); xr[4 + c] = f2bf(f1[c]); }
    xr[8] = f2bf(pnx); xr[9] = f2bf(pny); xr[10] = f2bf(pnz);
    #pragma unroll
    for (int c = 11; c < 32; ++c) xr[c] = f2bf(0.f);
  }
  __syncthreads();

  mlp_layer(Xs, 32,   img + OFF_WS1F, bimg + BOF_SB1, H1, HID_, 4, 8, true,  wave, lane, 8);
  __syncthreads();
  mlp_layer(H1, HID_, img + OFF_WS2F, bimg + BOF_SB2, H2, HID_, 4, 8, true,  wave, lane, 8);
  __syncthreads();
  mlp_layer(H2, HID_, img + OFF_WS3F, bimg + BOF_SB3, H3, 16,   4, 1, false, wave, lane, 8);
  __syncthreads();

  if (tid < 64) sdf_c[p] = inMask ? bf2f(H3[tid * 16]) : 0.f;
}

// ===========================================================================
// Kernel 2: NeuS weights + sample_pdf + merge-sort (one thread per ray)
// ===========================================================================
__global__ void __launch_bounds__(256)
FFS_sample(const float* __restrict__ z_vals, const float* __restrict__ sdf_c,
           float* __restrict__ dists_o, float* __restrict__ zmid_o) {
  const int r = blockIdx.x * blockDim.x + threadIdx.x;
  if (r >= R_) return;
  float z[NS_], sd[NS_], wp[NS_ - 1], cdf[NS_], zs[NI_], zm[NT_];
  for (int i = 0; i < NS_; ++i) { z[i] = z_vals[r * NS_ + i]; sd[i] = sdf_c[r * NS_ + i]; }

  float T = 1.f, prev_raw = 0.f, wsum = 0.f;
  for (int i = 0; i < NS_ - 1; ++i) {
    float dz  = z[i + 1] - z[i];
    float raw = (sd[i + 1] - sd[i]) / (dz + 1e-5f);
    float cc  = fminf(fmaxf(fminf(prev_raw, raw), -1000.f), 0.f);
    prev_raw  = raw;
    float mid = 0.5f * (sd[i] + sd[i + 1]);
    float pc  = sigm((mid - cc * dz * 0.5f) * 64.f);
    float nc  = sigm((mid + cc * dz * 0.5f) * 64.f);
    float a   = fminf(fmaxf((pc - nc + 1e-5f) / (pc + 1e-5f), 0.f), 1.f);
    float w   = a * T;
    T *= (1.f - a + 1e-7f);
    wp[i] = w + 1e-5f;
    wsum += wp[i];
  }
  cdf[0] = 0.f;
  for (int i = 1; i < NS_; ++i) cdf[i] = cdf[i - 1] + wp[i - 1] / wsum;

  const float lo = 0.5f / (float)NI_, hi = 1.f - 0.5f / (float)NI_;
  int idx = 0;
  for (int j = 0; j < NI_; ++j) {
    float u = lo + (hi - lo) * (float)j / (float)(NI_ - 1);
    while (idx < NS_ && cdf[idx] <= u) ++idx;          // searchsorted(side='right')
    int below = imax(idx - 1, 0), above = imin(idx, NS_ - 1);
    float cb = cdf[below], ca = cdf[above];
    float den = (ca - cb < 1e-5f) ? 1.f : (ca - cb);
    float t = (u - cb) / den;
    zs[j] = z[below] + t * (z[above] - z[below]);
  }
  int i = 0, j = 0;
  for (int k = 0; k < NT_; ++k) {
    bool takeZ = (j >= NI_) || (i < NS_ && z[i] <= zs[j]);
    zm[k] = takeZ ? z[i++] : zs[j++];
  }
  const float sdist = (FARP - NEARP) / (float)NS_;
  for (int k = 0; k < NT_; ++k) {
    float d = (k < NT_ - 1) ? (zm[k + 1] - zm[k]) : sdist;
    dists_o[r * NT_ + k] = d;
    zmid_o[r * NT_ + k]  = zm[k] + 0.5f * d;
  }
}

// ===========================================================================
// Kernel 3: fine pass — sdf fwd+vjp and rgb fwd, all GEMMs via WMMA
// ===========================================================================
__global__ void __launch_bounds__(256)
FFS_fine(const float* __restrict__ rays_o, const float* __restrict__ rays_d,
         const float* __restrict__ grid0, const float* __restrict__ grid1,
         const float* __restrict__ origin, const float* __restrict__ dims,
         const bf16t* __restrict__ wimg,
         const float* __restrict__ zmid,
         float* __restrict__ fsdf, float* __restrict__ fmask,
         float* __restrict__ fgrad, float* __restrict__ frgb) {
  extern __shared__ __align__(16) char smem[];
  bf16t* img  = (bf16t*)smem;
  float* bimg = (float*)(smem + BIAS_BYTE);
  bf16t* ap   = (bf16t*)(smem + IMG_PAD_BYTES);
  bf16t* Xs = ap; ap += 64 * 32;
  bf16t* H1 = ap; ap += 64 * HID_;
  bf16t* H2 = ap; ap += 64 * HID_;
  bf16t* H3 = ap; ap += 64 * 16;
  bf16t* Xr = ap; ap += 64 * 32;
  bf16t* R1 = ap; ap += 64 * HID_;
  bf16t* R2 = ap; ap += 64 * HID_;
  bf16t* R3 = ap; ap += 64 * 16;
  bf16t* G2 = ap; ap += 64 * HID_;
  bf16t* G1 = ap; ap += 64 * HID_;
  bf16t* GX = ap;

  const int tid = threadIdx.x;
  const int wave = tid >> 5, lane = tid & 31;
  stage_image(wimg, smem, wave, lane, 8);

  // ---- phase 1: per-point gather + build MLP inputs (state kept in registers)
  Tri t0{}, t1{};
  float invDx = 0, invDy = 0, invDz = 0;
  int inMask = 0;
  const int p = blockIdx.x * 64 + tid;
  if (tid < 64) {
    int ray = p >> 7;
    float dxr = rays_d[ray * 3 + 0], dyr = rays_d[ray * 3 + 1], dzr = rays_d[ray * 3 + 2];
    float zc = zmid[p];
    float px = rays_o[ray * 3 + 0] + dxr * zc;
    float py = rays_o[ray * 3 + 1] + dyr * zc;
    float pz = rays_o[ray * 3 + 2] + dzr * zc;
    float D0 = dims[0], D1 = dims[1], D2 = dims[2];
    invDx = 2.f / D0; invDy = 2.f / D1; invDz = 2.f / D2;
    float pnx = 2.f * (px - origin[0]) / D0 - 1.f;
    float pny = 2.f * (py - origin[1]) / D1 - 1.f;
    float pnz = 2.f * (pz - origin[2]) / D2 - 1.f;
    inMask = (fabsf(pnx) <= 1.f && fabsf(pny) <= 1.f && fabsf(pnz) <= 1.f) ? 1 : 0;
    float dn = sqrtf(dxr * dxr + dyr * dyr + dzr * dzr);
    float vdx = dxr / dn, vdy = dyr / dn, vdz = dzr / dn;
    t0 = tri_setup(pnx, pny, pnz, 96, 96, 96);
    t1 = tri_setup(pnx, pny, pnz, 192, 192, 192);
    __builtin_prefetch(grid0 + (((long)t0.x0 * 96 + t0.y0) * 96 + t0.z0), 0, 0);
    __builtin_prefetch(grid1 + (((long)t1.x0 * 192 + t1.y0) * 192 + t1.z0), 0, 0);
    float f0[8], f1[8];
    tri_eval<false, 8>(grid0, 96, 96, 96, t0, f0, nullptr, nullptr, nullptr);
    tri_eval<false, 8>(grid1, 192, 192, 192, t1, f1, nullptr, nullptr, nullptr);
    bf16t* xs = Xs + tid * 32;
    #pragma unroll
    for (int c = 0; c < 4; ++c) { xs[c] = f2bf(f0[c]); xs[4 + c] = f2bf(f1[c]); }
    xs[8] = f2bf(pnx); xs[9] = f2bf(pny); xs[10] = f2bf(pnz);
    #pragma unroll
    for (int c = 11; c < 32; ++c) xs[c] = f2bf(0.f);
    bf16t* xq = Xr + tid * 32;
    float ms = inMask ? 1.f : 0.f;
    #pragma unroll
    for (int c = 0; c < 4; ++c) { xq[c] = f2bf(f0[4 + c] * ms); xq[4 + c] = f2bf(f1[4 + c] * ms); }
    xq[8] = f2bf(vdx); xq[9] = f2bf(vdy); xq[10] = f2bf(vdz);
    xq[11] = f2bf(pnx); xq[12] = f2bf(pny); xq[13] = f2bf(pnz);
    #pragma unroll
    for (int c = 14; c < 32; ++c) xq[c] = f2bf(0.f);
  }
  __syncthreads();

  // sdf forward
  mlp_layer(Xs, 32,   img + OFF_WS1F, bimg + BOF_SB1, H1, HID_, 4, 8, true,  wave, lane, 8);
  __syncthreads();
  mlp_layer(H1, HID_, img + OFF_WS2F, bimg + BOF_SB2, H2, HID_, 4, 8, true,  wave, lane, 8);
  __syncthreads();
  mlp_layer(H2, HID_, img + OFF_WS3F, bimg + BOF_SB3, H3, 16,   4, 1, false, wave, lane, 8);
  __syncthreads();
  // sdf backward: g2 = W3 .* relu'(h2);  W3[j] lives in the Ws3f fragment image
  for (int i = tid; i < 64 * HID_; i += 256) {
    int j = i & (HID_ - 1);
    float w3 = bf2f(img[OFF_WS3F + ((j >> 5) << 9) + (((j >> 4) & 1) << 8) + (j & 15)]);
    G2[i] = f2bf(bf2f(H2[i]) > 0.f ? w3 : 0.f);
  }
  __syncthreads();
  mlp_layer(G2, HID_, img + OFF_WS2T, nullptr, G1, HID_, 4, 8, false, wave, lane, 8);
  __syncthreads();
  for (int i = tid; i < 64 * HID_; i += 256)
    if (!(bf2f(H1[i]) > 0.f)) G1[i] = f2bf(0.f);
  __syncthreads();
  mlp_layer(G1, HID_, img + OFF_WS1T, nullptr, GX, 16, 4, 1, false, wave, lane, 8);
  __syncthreads();
  // rgb forward
  mlp_layer(Xr, 32,   img + OFF_RS1F, bimg + BOF_RB1, R1, HID_, 4, 8, true,  wave, lane, 8);
  __syncthreads();
  mlp_layer(R1, HID_, img + OFF_RS2F, bimg + BOF_RB2, R2, HID_, 4, 8, true,  wave, lane, 8);
  __syncthreads();
  mlp_layer(R2, HID_, img + OFF_RS3F, bimg + BOF_RB3, R3, 16,   4, 1, false, wave, lane, 8);
  __syncthreads();

  // ---- phase 2: chain grads through trilinear, write outputs
  if (tid < 64) {
    float gf0[4], gf1[4];
    #pragma unroll
    for (int c = 0; c < 4; ++c) { gf0[c] = bf2f(GX[tid * 16 + c]); gf1[c] = bf2f(GX[tid * 16 + 4 + c]); }
    float gpx = bf2f(GX[tid * 16 + 8]);
    float gpy = bf2f(GX[tid * 16 + 9]);
    float gpz = bf2f(GX[tid * 16 + 10]);
    float f0[4], ax0[4], ay0[4], az0[4];
    float f1[4], ax1[4], ay1[4], az1[4];
    tri_eval<true, 4>(grid0, 96, 96, 96, t0, f0, ax0, ay0, az0);
    tri_eval<true, 4>(grid1, 192, 192, 192, t1, f1, ax1, ay1, az1);
    float sx = 0.f, sy = 0.f, sz = 0.f;
    #pragma unroll
    for (int c = 0; c < 4; ++c) {
      sx += gf0[c] * ax0[c] * t0.dcx + gf1[c] * ax1[c] * t1.dcx;
      sy += gf0[c] * ay0[c] * t0.dcy + gf1[c] * ay1[c] * t1.dcy;
      sz += gf0[c] * az0[c] * t0.dcz + gf1[c] * az1[c] * t1.dcz;
    }
    float gx = (sx + gpx) * invDx;
    float gy = (sy + gpy) * invDy;
    float gz = (sz + gpz) * invDz;
    if (!inMask) { gx = 0.f; gy = 0.f; gz = 0.f; }
    fsdf[p]  = inMask ? bf2f(H3[tid * 16]) : 0.f;
    fmask[p] = inMask ? 1.f : 0.f;
    fgrad[p * 3 + 0] = gx; fgrad[p * 3 + 1] = gy; fgrad[p * 3 + 2] = gz;
    frgb[p * 3 + 0] = sigm(bf2f(R3[tid * 16 + 0]));
    frgb[p * 3 + 1] = sigm(bf2f(R3[tid * 16 + 1]));
    frgb[p * 3 + 2] = sigm(bf2f(R3[tid * 16 + 2]));
  }
}

// ===========================================================================
// Kernel 4: composite along each ray (one thread per ray)
// ===========================================================================
__global__ void __launch_bounds__(256)
FFS_composite(const float* __restrict__ rays_d,
              const float* __restrict__ fsdf, const float* __restrict__ fmask,
              const float* __restrict__ fgrad, const float* __restrict__ frgb,
              const float* __restrict__ dists, const float* __restrict__ zmid,
              float* __restrict__ out) {
  const int r = blockIdx.x * blockDim.x + threadIdx.x;
  if (r >= R_) return;
  float dx = rays_d[r * 3 + 0], dy = rays_d[r * 3 + 1], dz = rays_d[r * 3 + 2];
  float dn = sqrtf(dx * dx + dy * dy + dz * dz);
  float vx = dx / dn, vy = dy / dn, vz = dz / dn;
  float T = 1.f, c0 = 0.f, c1 = 0.f, c2 = 0.f, dep = 0.f, g0 = 0.f, g1 = 0.f, g2 = 0.f;
  for (int i = 0; i < NT_; ++i) {
    int idx = r * NT_ + i;
    float s = fsdf[idx];
    float gx = fgrad[idx * 3 + 0], gy = fgrad[idx * 3 + 1], gz = fgrad[idx * 3 + 2];
    float cosv = fminf(vx * gx + vy * gy + vz * gz, 0.f);  // -relu(-dot)
    float d = dists[idx];
    float pc = sigm((s - cosv * d * 0.5f) * 64.f);
    float nc = sigm((s + cosv * d * 0.5f) * 64.f);
    float a  = fminf(fmaxf((pc - nc + 1e-5f) / (pc + 1e-5f), 0.f), 1.f);
    float w  = a * T;
    T *= (1.f - a + 1e-7f);
    w = (fmask[idx] > 0.5f) ? w : 0.f;
    c0 += w * frgb[idx * 3 + 0]; c1 += w * frgb[idx * 3 + 1]; c2 += w * frgb[idx * 3 + 2];
    dep += w * zmid[idx];
    g0 += w * gx; g1 += w * gy; g2 += w * gz;
  }
  float gn = sqrtf(g0 * g0 + g1 * g1 + g2 * g2) + 1e-8f;
  out[r * 7 + 0] = c0; out[r * 7 + 1] = c1; out[r * 7 + 2] = c2;
  out[r * 7 + 3] = dep;
  out[r * 7 + 4] = g0 / gn; out[r * 7 + 5] = g1 / gn; out[r * 7 + 6] = g2 / gn;
}

// ===========================================================================
// Host-side launcher
// ===========================================================================
static constexpr size_t COARSE_SMEM =
    (size_t)IMG_PAD_BYTES + (size_t)(64 * 32 + 64 * 128 + 64 * 128 + 64 * 16) * 2;
static constexpr size_t FINE_SMEM =
    (size_t)IMG_PAD_BYTES +
    (size_t)(2 * (64 * 32 + 64 * 128 + 64 * 128 + 64 * 16) +
             (64 * 128 + 64 * 128 + 64 * 16)) * 2;

extern "C" void kernel_launch(void* const* d_in, const int* in_sizes, int n_in,
                              void* d_out, int out_size, void* d_ws, size_t ws_size,
                              hipStream_t stream) {
  (void)in_sizes; (void)n_in; (void)out_size; (void)ws_size;
  const float* rays_o = (const float*)d_in[0];
  const float* rays_d = (const float*)d_in[1];
  const float* z_vals = (const float*)d_in[2];
  const float* grid0  = (const float*)d_in[3];
  const float* grid1  = (const float*)d_in[4];
  const float* origin = (const float*)d_in[5];
  const float* dims   = (const float*)d_in[6];
  const float* sW1 = (const float*)d_in[7];
  const float* sb1 = (const float*)d_in[8];
  const float* sW2 = (const float*)d_in[9];
  const float* sb2 = (const float*)d_in[10];
  const float* sW3 = (const float*)d_in[11];
  const float* sb3 = (const float*)d_in[12];
  const float* rW1 = (const float*)d_in[13];
  const float* rb1 = (const float*)d_in[14];
  const float* rW2 = (const float*)d_in[15];
  const float* rb2 = (const float*)d_in[16];
  const float* rW3 = (const float*)d_in[17];
  const float* rb3 = (const float*)d_in[18];

  bf16t* wimg  = (bf16t*)d_ws;                          // IMG_PAD_BYTES
  float* ws    = (float*)((char*)d_ws + IMG_PAD_BYTES);
  float* sdf_c = ws;                                    // R*NS
  float* dists = sdf_c + (size_t)R_ * NS_;              // R*NT
  float* zmid  = dists + (size_t)R_ * NT_;
  float* fsdf  = zmid  + (size_t)R_ * NT_;
  float* fmask = fsdf  + (size_t)R_ * NT_;
  float* fgrad = fmask + (size_t)R_ * NT_;              // R*NT*3
  float* frgb  = fgrad + (size_t)R_ * NT_ * 3;

  dim3 blk(256);
  FFS_prep<<<32, blk, 0, stream>>>(sW1, sb1, sW2, sb2, sW3, sb3,
                                   rW1, rb1, rW2, rb2, rW3, rb3, wimg);
  FFS_coarse<<<(R_ * NS_) / 64, blk, COARSE_SMEM, stream>>>(
      rays_o, rays_d, z_vals, grid0, grid1, origin, dims, wimg, sdf_c);
  FFS_sample<<<(R_ + 255) / 256, blk, 0, stream>>>(z_vals, sdf_c, dists, zmid);
  FFS_fine<<<(R_ * NT_) / 64, blk, FINE_SMEM, stream>>>(
      rays_o, rays_d, grid0, grid1, origin, dims, wimg,
      zmid, fsdf, fmask, fgrad, frgb);
  FFS_composite<<<(R_ + 255) / 256, blk, 0, stream>>>(
      rays_d, fsdf, fmask, fgrad, frgb, dists, zmid, (float*)d_out);
}
